// MLP_GNN_57604101374612
// MI455X (gfx1250) — matrix operand order repaired and checked by reading, not compiled, not run on hardware
//
#include <hip/hip_runtime.h>
#include <hip/hip_fp16.h>

// ---------------------------------------------------------------------------
// MI455X (gfx1250, wave32) implementation of the MLP+GCN reference.
//  - GEMMs run on v_wmma_f32_16x16x32_f16 (f32 accumulate).
//  - Weights pre-swizzled into exact B-fragment layout (1 contiguous 32B
//    load per lane per fragment).
//  - A tiles staged into LDS via the Tensor Data Mover (tensor_load_to_lds,
//    TENSORcnt-tracked), fragments gathered with ds_load_b128 pairs per the
//    CDNA5 16-bit A-matrix VGPR layout.
//  - Edge aggregation: coalesced gather + f32 atomic scatter (mem-bound,
//    ~70us/conv at 23.3 TB/s) with global_prefetch of the next source row.
// Assumes N % 16 == 0 (reference: N = 50000).
// ---------------------------------------------------------------------------

typedef __attribute__((ext_vector_type(16))) _Float16 v16h;
typedef __attribute__((ext_vector_type(8)))  _Float16 v8h;
typedef __attribute__((ext_vector_type(8)))  float    v8f;
typedef __attribute__((ext_vector_type(4)))  unsigned int u32x4;
typedef __attribute__((ext_vector_type(8)))  unsigned int u32x8;

static constexpr int HD = 256;   // hidden dim
static constexpr int IN_DIM = 26;

#define USE_TDM 1   // 1: stage A tile with tensor_load_to_lds ; 0: vector copy

// ----------------------------- utility kernels -----------------------------

__global__ __launch_bounds__(256) void k_zero(float* __restrict__ p, long long cnt) {
  long long i = (long long)blockIdx.x * 256 + threadIdx.x;
  if (i < cnt) p[i] = 0.f;
}

// x [N,26] f32 -> Xb [N,32] f16 (K padded with zeros)
__global__ __launch_bounds__(256) void k_pack_x(const float* __restrict__ x,
                                                _Float16* __restrict__ xb, int n) {
  int idx = blockIdx.x * 256 + threadIdx.x;
  int r = idx >> 5, c = idx & 31;
  if (r < n) xb[idx] = (_Float16)((c < IN_DIM) ? x[r * IN_DIM + c] : 0.f);
}

// W [Ksrc,256] f32 row-major -> packed f16 B fragments.
// Fragment (nt,kt) holds B tile K=[kt*32,kt*32+32) x N=[nt*16,nt*16+16):
//   lane l: N = l&15, half = l>>4 ; element j(0..15): K = kt*32 + (j<8 ? half*8+j : 16+half*8+(j-8))
// stored contiguously: bp[(nt*KT+kt)*512 + l*16 + j]
__global__ __launch_bounds__(256) void k_pack_w(const float* __restrict__ w,
                                                _Float16* __restrict__ bp,
                                                int ksrc, int kt_cnt) {
  int idx = blockIdx.x * 256 + threadIdx.x;
  if (idx >= kt_cnt * 16 * 512) return;
  int j    = idx & 15;
  int lane = (idx >> 4) & 31;
  int frag = idx >> 9;
  int kt = frag % kt_cnt;
  int nt = frag / kt_cnt;
  int n = nt * 16 + (lane & 15);
  int k = kt * 32 + (lane >> 4) * 8 + (j & 7) + ((j >> 3) << 4);
  bp[idx] = (_Float16)((k < ksrc) ? w[k * HD + n] : 0.f);
}

// ------------------------------- WMMA GEMM ---------------------------------
// C[N,256] = A[N,K] @ W + bias ; optional relu ; f32 and/or f16 outputs.
// Block: 16-row tile, 256 threads = 8 waves, each wave owns 2 n-tiles.
__global__ __launch_bounds__(256) void k_gemm(
    const _Float16* __restrict__ A, const _Float16* __restrict__ Bp,
    const float* __restrict__ bias, float* __restrict__ outF,
    _Float16* __restrict__ outH, int K, int doRelu) {
  __shared__ __align__(16) _Float16 Asm[16 * 256];
  const int KT = K >> 5;
  const int m0 = blockIdx.x << 4;
  const int lane = threadIdx.x & 31;
  const int wv   = threadIdx.x >> 5;

#if USE_TDM
  {
    // Each wave DMAs its 2-row slice of the A tile (2*K f16 = K dwords,
    // contiguous in memory and LDS) via the Tensor Data Mover.
    // D# group0: count=1 | lds_addr | 57-bit global addr | type=2.
    // D# group1: data_size=4B, 1-D tile: tensor_dim0 = tile_dim0 = K dwords.
    const unsigned ndw = (unsigned)K;  // dwords in 2 rows
    const unsigned long long ga =
        (unsigned long long)(const void*)(A + (size_t)(m0 + wv * 2) * K);
    typedef __attribute__((address_space(3))) const void* lds_cptr_t;
    const unsigned ldsb = (unsigned)(size_t)(lds_cptr_t)(Asm + wv * 2 * K);
    u32x4 g0;
    g0[0] = 1u;                                    // count=1 (valid user D#)
    g0[1] = ldsb;                                  // lds_addr (bytes)
    g0[2] = (unsigned)(ga & 0xffffffffu);          // global_addr[31:0]
    g0[3] = ((unsigned)(ga >> 32) & 0x01ffffffu)   // global_addr[56:32]
            | (2u << 30);                          // type = 2 ("image")
    u32x8 g1;
    g1[0] = (2u << 16);                            // data_size = 4 bytes
    g1[1] = (ndw & 0xffffu) << 16;                 // tensor_dim0[15:0]
    g1[2] = ((ndw >> 16) & 0xffffu) | (1u << 16);  // tensor_dim0 hi | tdim1=1
    g1[3] = (ndw & 0xffffu) << 16;                 // tile_dim0 = ndw
    g1[4] = 0u;                                    // tile_dim1/2 unused
    g1[5] = ndw;                                   // tensor_dim0_stride
    g1[6] = 0u;
    g1[7] = 0u;
    asm volatile("tensor_load_to_lds %0, %1" :: "s"(g0), "s"(g1) : "memory");
    __builtin_amdgcn_s_wait_tensorcnt(0);
  }
  __syncthreads();
#else
  {
    // Whole A tile (16 rows x K) is contiguous in memory; coalesced b128 copy.
    const uint4* src = (const uint4*)(A + (size_t)m0 * K);
    uint4* dst = (uint4*)Asm;
    const int chunks = (16 * K) >> 3;  // 16B chunks
    for (int i = threadIdx.x; i < chunks; i += 256) dst[i] = src[i];
  }
  __syncthreads();
#endif

  const int hf   = lane >> 4;   // lane half
  const int lrow = lane & 15;
  const int nt0 = wv * 2, nt1 = wv * 2 + 1;

  v8f acc0 = {};
  v8f acc1 = {};
  for (int kt = 0; kt < KT; ++kt) {
    // A fragment per CDNA5 16-bit 16x32 layout:
    //   lane l (M=l&15, half=l>>4): j<8 -> K=j+half*8 ; j>=8 -> K=16+(j-8)+half*8
    // => two contiguous 8-element (16B) chunks -> 2x ds_load_b128.
    const _Float16* ap = Asm + lrow * K + (kt << 5) + hf * 8;
    v8h alo = *(const v8h*)(ap);
    v8h ahi = *(const v8h*)(ap + 16);
    v16h a = __builtin_shufflevector(alo, ahi, 0, 1, 2, 3, 4, 5, 6, 7,
                                     8, 9, 10, 11, 12, 13, 14, 15);
    v16h b0 = *(const v16h*)(Bp + (((size_t)(nt0 * KT + kt) << 9) + ((size_t)lane << 4)));
    v16h b1 = *(const v16h*)(Bp + (((size_t)(nt1 * KT + kt) << 9) + ((size_t)lane << 4)));
    acc0 = __builtin_amdgcn_wmma_f32_16x16x32_f16(false, a, false, b0,
                                                  (short)0, acc0, false, false);
    acc1 = __builtin_amdgcn_wmma_f32_16x16x32_f16(false, a, false, b1,
                                                  (short)0, acc1, false, false);
  }

  // C/D layout: lane half hf, col = lane&15, VGPR v -> row M = v + hf*8.
  const float bv0 = bias ? bias[nt0 * 16 + lrow] : 0.f;
  const float bv1 = bias ? bias[nt1 * 16 + lrow] : 0.f;
#pragma unroll
  for (int v = 0; v < 8; ++v) {
    const int rowg = m0 + hf * 8 + v;
    float x0 = acc0[v] + bv0;
    float x1 = acc1[v] + bv1;
    if (doRelu) { x0 = fmaxf(x0, 0.f); x1 = fmaxf(x1, 0.f); }
    const size_t o0 = (size_t)rowg * HD + nt0 * 16 + lrow;
    const size_t o1 = (size_t)rowg * HD + nt1 * 16 + lrow;
    if (outF) { outF[o0] = x0; outF[o1] = x1; }
    if (outH) { outH[o0] = (_Float16)x0; outH[o1] = (_Float16)x1; }
  }
}

// ------------------------------ GCN support --------------------------------

__global__ __launch_bounds__(256) void k_deg(const long long* __restrict__ col,
                                             float* __restrict__ deg, int E) {
  int e = blockIdx.x * 256 + threadIdx.x;
  if (e < E) atomicAdd(&deg[(int)col[e]], 1.0f);
}

__global__ __launch_bounds__(256) void k_deg_fin(float* __restrict__ dinv, int n) {
  int i = blockIdx.x * 256 + threadIdx.x;
  if (i < n) dinv[i] = rsqrtf(dinv[i] + 1.0f);  // +1 self loop
}

// out[col[e]] += HW[row[e]] * dinv[row]*dinv[col] ; one wave per edge,
// each lane moves 8 channels (2x b128 gather + 8 f32 atomics).
__global__ __launch_bounds__(256) void k_scatter(
    const float* __restrict__ HW, const long long* __restrict__ row,
    const long long* __restrict__ col, const float* __restrict__ dinv,
    float* __restrict__ out, int E) {
  const int lane = threadIdx.x & 31;
  const int e = blockIdx.x * 8 + (threadIdx.x >> 5);
  if (e >= E) return;
  const int s = (int)row[e];
  const int t = (int)col[e];
  if (e + 8 < E) {  // prefetch next edge's source row (global_prefetch_b8)
    const int sn = (int)row[e + 8];
    __builtin_prefetch(HW + (size_t)sn * HD + lane * 8, 0, 1);
  }
  const float nrm = dinv[s] * dinv[t];
  const float4* src = (const float4*)(HW + (size_t)s * HD) + (lane << 1);
  const float4 a = src[0];
  const float4 b = src[1];
  float* dst = out + (size_t)t * HD + (lane << 3);
  atomicAdd(dst + 0, a.x * nrm); atomicAdd(dst + 1, a.y * nrm);
  atomicAdd(dst + 2, a.z * nrm); atomicAdd(dst + 3, a.w * nrm);
  atomicAdd(dst + 4, b.x * nrm); atomicAdd(dst + 5, b.y * nrm);
  atomicAdd(dst + 6, b.z * nrm); atomicAdd(dst + 7, b.w * nrm);
}

// agg += HW * (1/deg)  (self-loop: dinv^2) + bias[c]
__global__ __launch_bounds__(256) void k_finish(float* __restrict__ agg,
                                                const float* __restrict__ HW,
                                                const float* __restrict__ dinv,
                                                const float* __restrict__ bias,
                                                int n) {
  int idx = blockIdx.x * 256 + threadIdx.x;
  if (idx >= n * HD) return;
  const int r = idx >> 8, c = idx & 255;
  const float di = dinv[r];
  agg[idx] = agg[idx] + HW[idx] * di * di + bias[c];
}

// ------------------------------- BatchNorm ---------------------------------
// stats: [0:256)=sum, [256:512)=sumsq, [512:768)=mean, [768:1024)=istd
__global__ __launch_bounds__(256) void k_bnstats(const float* __restrict__ h,
                                                 float* __restrict__ stats, int n) {
  const int c = threadIdx.x;
  const int r0 = blockIdx.x * 256;
  float s = 0.f, s2 = 0.f;
  for (int r = 0; r < 256; ++r) {
    if (r0 + r >= n) break;
    const float v = h[(size_t)(r0 + r) * HD + c];
    s += v; s2 += v * v;
  }
  atomicAdd(&stats[c], s);
  atomicAdd(&stats[256 + c], s2);
}

__global__ __launch_bounds__(256) void k_bnfin(float* __restrict__ stats, int n) {
  const int c = threadIdx.x;
  const float m = stats[c] / (float)n;
  const float var = stats[256 + c] / (float)n - m * m;
  stats[512 + c] = m;
  stats[768 + c] = rsqrtf(var + 1e-5f);
}

// G = relu(bn1(agg)) + G (residual, in place) ; Gh = f16(G)
__global__ __launch_bounds__(256) void k_bnapply_res(
    const float* __restrict__ agg, const float* __restrict__ stats,
    const float* __restrict__ gamma, const float* __restrict__ beta,
    float* __restrict__ G, _Float16* __restrict__ Gh, int n) {
  int idx = blockIdx.x * 256 + threadIdx.x;
  if (idx >= n * HD) return;
  const int c = idx & 255;
  float hv = gamma[c] * (agg[idx] - stats[512 + c]) * stats[768 + c] + beta[c];
  hv = fmaxf(hv, 0.f);
  const float g = hv + G[idx];
  G[idx] = g;
  Gh[idx] = (_Float16)g;
}

// y[i] = bn2(h[i,:]) . w_out + b_out ; one wave per node, shfl reduction
__global__ __launch_bounds__(256) void k_out(
    const float* __restrict__ h, const float* __restrict__ stats,
    const float* __restrict__ gamma, const float* __restrict__ beta,
    const float* __restrict__ wo, const float* __restrict__ bo,
    float* __restrict__ y, int n) {
  const int lane = threadIdx.x & 31;
  const int node = blockIdx.x * 8 + (threadIdx.x >> 5);
  if (node >= n) return;
  float s = 0.f;
#pragma unroll
  for (int i = 0; i < 8; ++i) {
    const int c = lane * 8 + i;
    float v = h[(size_t)node * HD + c];
    v = gamma[c] * (v - stats[512 + c]) * stats[768 + c] + beta[c];
    s += v * wo[c];
  }
#pragma unroll
  for (int off = 16; off > 0; off >>= 1) s += __shfl_down(s, off, 32);
  if (lane == 0) y[node] = s + bo[0];
}

// ------------------------------- launcher ----------------------------------

extern "C" void kernel_launch(void* const* d_in, const int* in_sizes, int n_in,
                              void* d_out, int out_size, void* d_ws, size_t ws_size,
                              hipStream_t stream) {
  const float*     x   = (const float*)d_in[0];
  const long long* ei  = (const long long*)d_in[1];  // int64 (2,E)
  const float* w1  = (const float*)d_in[3];
  const float* b1  = (const float*)d_in[4];
  const float* w2  = (const float*)d_in[5];
  const float* b2  = (const float*)d_in[6];
  const float* wc1 = (const float*)d_in[7];
  const float* bc1 = (const float*)d_in[8];
  const float* wc2 = (const float*)d_in[9];
  const float* bc2 = (const float*)d_in[10];
  const float* g1  = (const float*)d_in[11];
  const float* be1 = (const float*)d_in[12];
  const float* g2  = (const float*)d_in[13];
  const float* be2 = (const float*)d_in[14];
  const float* wo  = (const float*)d_in[15];
  const float* bo  = (const float*)d_in[16];

  const int N = in_sizes[0] / IN_DIM;
  const int E = in_sizes[1] / 2;
  const long long* row = ei;
  const long long* col = ei + E;

  // workspace carve-out (256B aligned)
  char* ws = (char*)d_ws;
  size_t off = 0;
  auto carve = [&](size_t bytes) -> char* {
    char* p = ws + off;
    off = (off + bytes + 255) & ~(size_t)255;
    return p;
  };
  _Float16* Xb   = (_Float16*)carve((size_t)N * 32 * 2);
  _Float16* W1p  = (_Float16*)carve((size_t)1 * 16 * 512 * 2);
  _Float16* W2p  = (_Float16*)carve((size_t)8 * 16 * 512 * 2);
  _Float16* Wc1p = (_Float16*)carve((size_t)8 * 16 * 512 * 2);
  _Float16* Wc2p = (_Float16*)carve((size_t)8 * 16 * 512 * 2);
  _Float16* B0   = (_Float16*)carve((size_t)N * HD * 2);  // h16 activations (ping)
  _Float16* B1   = (_Float16*)carve((size_t)N * HD * 2);  // h16 activations (pong)
  float* F0   = (float*)carve((size_t)N * HD * 4);  // gnn_input (residual)
  float* F1   = (float*)carve((size_t)N * HD * 4);  // h@W (conv messages)
  float* F2   = (float*)carve((size_t)N * HD * 4);  // aggregation / conv out
  float* dinv = (float*)carve((size_t)N * 4);
  float* stats = (float*)carve(1024 * 4);
  (void)ws_size; (void)n_in; (void)out_size;

  const dim3 blk(256);
  const int gNH  = (N * HD + 255) / 256;
  const int gN   = (N + 255) / 256;
  const int gE   = (E + 255) / 256;
  const int gTile = N / 16;  // N % 16 == 0

  // pack inputs + weights into WMMA-friendly layouts
  k_pack_x<<<(N * 32 + 255) / 256, blk, 0, stream>>>(x, Xb, N);
  k_pack_w<<<(1 * 16 * 512 + 255) / 256, blk, 0, stream>>>(w1, W1p, IN_DIM, 1);
  k_pack_w<<<(8 * 16 * 512 + 255) / 256, blk, 0, stream>>>(w2, W2p, HD, 8);
  k_pack_w<<<(8 * 16 * 512 + 255) / 256, blk, 0, stream>>>(wc1, Wc1p, HD, 8);
  k_pack_w<<<(8 * 16 * 512 + 255) / 256, blk, 0, stream>>>(wc2, Wc2p, HD, 8);

  // symmetric degree normalization (shared by both convs)
  k_zero<<<gN, blk, 0, stream>>>(dinv, N);
  k_deg<<<gE, blk, 0, stream>>>(col, dinv, E);
  k_deg_fin<<<gN, blk, 0, stream>>>(dinv, N);

  // MLP: B0 = relu(Xb @ W1 + b1) ; F0 = gnn_input = B0 @ W2 + b2 (+f16 copy B1)
  k_gemm<<<gTile, blk, 0, stream>>>(Xb, W1p, b1, nullptr, B0, 32, 1);
  k_gemm<<<gTile, blk, 0, stream>>>(B0, W2p, b2, F0, B1, HD, 0);

  // conv1: F1 = gnn_input @ Wc1 ; aggregate into F2 ; + self-loop + bias
  k_gemm<<<gTile, blk, 0, stream>>>(B1, Wc1p, nullptr, F1, nullptr, HD, 0);
  k_zero<<<gNH, blk, 0, stream>>>(F2, (long long)N * HD);
  k_scatter<<<(E + 7) / 8, blk, 0, stream>>>(F1, row, col, dinv, F2, E);
  k_finish<<<gNH, blk, 0, stream>>>(F2, F1, dinv, bc1, N);

  // bn1 + relu + residual -> F0 (and f16 copy in B0)
  k_zero<<<4, blk, 0, stream>>>(stats, 1024);
  k_bnstats<<<gN, blk, 0, stream>>>(F2, stats, N);
  k_bnfin<<<1, blk, 0, stream>>>(stats, N);
  k_bnapply_res<<<gNH, blk, 0, stream>>>(F2, stats, g1, be1, F0, B0, N);

  // conv2
  k_gemm<<<gTile, blk, 0, stream>>>(B0, Wc2p, nullptr, F1, nullptr, HD, 0);
  k_zero<<<gNH, blk, 0, stream>>>(F2, (long long)N * HD);
  k_scatter<<<(E + 7) / 8, blk, 0, stream>>>(F1, row, col, dinv, F2, E);
  k_finish<<<gNH, blk, 0, stream>>>(F2, F1, dinv, bc2, N);

  // bn2 + output projection
  k_zero<<<4, blk, 0, stream>>>(stats, 1024);
  k_bnstats<<<gN, blk, 0, stream>>>(F2, stats, N);
  k_bnfin<<<1, blk, 0, stream>>>(stats, N);
  k_out<<<(N + 7) / 8, blk, 0, stream>>>(F2, stats, g2, be2, wo, bo,
                                         (float*)d_out, N);
}
